// AttentionPooling_76029511074168
// MI455X (gfx1250) — matrix operand order repaired and checked by reading, not compiled, not run on hardware
//
#include <hip/hip_runtime.h>
#include <hip/hip_bf16.h>

// ---------------- problem constants ----------------
#define B_   32
#define NSEQ 4096
#define D_   768
#define H_   12
#define HP_  16          // heads padded to 16 for WMMA N dim
#define HD_  64
#define SCALE_ 0.125f    // 1/sqrt(64)

#define SPLITS 8                         // blocks per batch for pass over N
#define ROWS   64                        // rows (n) per chunk staged in LDS
#define CHUNKS (NSEQ / SPLITS / ROWS)    // 8
#define XPITCH (D_ + 16)                 // bf16 LDS row pitch (bank skew)

typedef __attribute__((ext_vector_type(16))) __bf16 v16bf;
typedef __attribute__((ext_vector_type(8)))  float  v8f;
struct BFrag { uint4 lo; uint4 hi; };    // 32B == sizeof(v16bf)

__device__ __forceinline__ unsigned short f32_to_bf16(float f) {
  unsigned u = __builtin_bit_cast(unsigned, f);
  unsigned r = 0x7FFFu + ((u >> 16) & 1u);     // round-to-nearest-even
  return (unsigned short)((u + r) >> 16);
}
__device__ __forceinline__ float bf16_to_f32(unsigned short h) {
  return __builtin_bit_cast(float, (unsigned)h << 16);
}

// ---------------------------------------------------------------------------
// Kernel 1 (tiny): q = latent@Wq + bq ; wq_eff[h][d] = Wk_h @ q_h (bf16);
// sq[h] = scale * (q_h . bk_h)
// ---------------------------------------------------------------------------
__global__ __launch_bounds__(256) void k1_prepare(
    const float* __restrict__ latent, const float* __restrict__ Wq,
    const float* __restrict__ bq, const float* __restrict__ Wkv,
    const float* __restrict__ bkv,
    unsigned short* __restrict__ wsWq,   // [16][768] bf16, head-major
    float* __restrict__ wsSq)            // [16]
{
  __shared__ float qs[D_];
  const int t = threadIdx.x;
  for (int i = t; i < D_; i += 256) {
    float s = bq[i];
    for (int d = 0; d < D_; ++d) s = fmaf(latent[d], Wq[(size_t)d * D_ + i], s);
    qs[i] = s;
  }
  __syncthreads();
  for (int o = t; o < HP_ * D_; o += 256) {
    int n = o / D_, d = o % D_;
    float v = 0.f;
    if (n < H_) {
      const float* wr = Wkv + (size_t)d * (2 * D_) + n * HD_;
      for (int j = 0; j < HD_; ++j) v = fmaf(wr[j], qs[n * HD_ + j], v);
    }
    wsWq[o] = f32_to_bf16(v);
  }
  if (t < HP_) {
    float s = 0.f;
    if (t < H_)
      for (int j = 0; j < HD_; ++j) s = fmaf(bkv[t * HD_ + j], qs[t * HD_ + j], s);
    wsSq[t] = s * SCALE_;
  }
}

// ---------------------------------------------------------------------------
// Kernel 2 (the streaming pass): single read of x.
// Per block: 512 rows of one batch. Chunks of 64 rows:
//   global fp32 -> LDS bf16 tile -> WMMA scores (waves 0-3) -> online softmax
//   stats -> xbar accumulation (registers) from the same LDS tile.
// ---------------------------------------------------------------------------
__global__ __launch_bounds__(256) void k2_scores_pool(
    const float* __restrict__ x,
    const unsigned short* __restrict__ wsWq, const float* __restrict__ wsSq,
    float* __restrict__ wsM, float* __restrict__ wsS,
    float* __restrict__ wsXb)
{
  __shared__ __attribute__((aligned(16))) unsigned short xt[ROWS][XPITCH];
  __shared__ __attribute__((aligned(16))) unsigned short wqT[HP_][XPITCH];
  __shared__ float scL[ROWS][17];
  __shared__ float awL[ROWS][17];
  __shared__ float mS[HP_], sS[HP_], fS[HP_], sqS[HP_];

  const int t = threadIdx.x;
  const int blk = blockIdx.x;
  const int b = blk / SPLITS, split = blk % SPLITS;
  const int wave = t >> 5, lane = t & 31, half = lane >> 4, lm = lane & 15;

  // stage wq_eff^T (bf16) into LDS; init online-softmax state
  for (int o = t; o < HP_ * D_; o += 256) wqT[o / D_][o % D_] = wsWq[o];
  if (t < HP_) {
    sqS[t] = wsSq[t];
    mS[t] = -__builtin_inff();
    sS[t] = 0.f;
    fS[t] = 0.f;
  }

  float acc[H_][3];
#pragma unroll
  for (int h = 0; h < H_; ++h)
#pragma unroll
    for (int j = 0; j < 3; ++j) acc[h][j] = 0.f;

  const size_t base = ((size_t)b * NSEQ + (size_t)split * (NSEQ / SPLITS)) * D_;

  for (int ch = 0; ch < CHUNKS; ++ch) {
    __syncthreads();  // previous chunk's tile fully consumed
    // ---- load 64 rows of x, convert fp32 -> bf16 into LDS ----
    const float4* xg = reinterpret_cast<const float4*>(x + base) +
                       (size_t)ch * (ROWS * D_ / 4);
    for (int i = 0; i < (ROWS * D_ / 4) / 256; ++i) {   // 48 iters
      int f4 = t + 256 * i;
      int row = f4 / (D_ / 4);
      int col = (f4 % (D_ / 4)) * 4;
      float4 v = xg[f4];
      if ((i & 7) == 0 && ch + 1 < CHUNKS)
        __builtin_prefetch(&xg[f4 + ROWS * D_ / 4], 0, 0);  // next chunk
      unsigned lo = (unsigned)f32_to_bf16(v.x) | ((unsigned)f32_to_bf16(v.y) << 16);
      unsigned hi = (unsigned)f32_to_bf16(v.z) | ((unsigned)f32_to_bf16(v.w) << 16);
      *reinterpret_cast<uint2*>(&xt[row][col]) = make_uint2(lo, hi);
    }
    __syncthreads();

    // ---- WMMA: scores tile (64 x 16) = xtile (64x768) @ wq_eff (768x16) ----
    if (wave < 4) {
      const int r0 = wave * 16;
      v8f c = {};
#pragma unroll
      for (int ks = 0; ks < D_ / 32; ++ks) {   // 24 WMMAs, K=32 each
        const int K0 = ks * 32;
        BFrag fa, fb;
        // A 16x32 bf16: lanes 0-15 hold K{0..7,16..23}, lanes 16-31 K{8..15,24..31}
        fa.lo = *reinterpret_cast<const uint4*>(&xt[r0 + lm][K0 + half * 8]);
        fa.hi = *reinterpret_cast<const uint4*>(&xt[r0 + lm][K0 + 16 + half * 8]);
        // B 32x16 bf16: lanes 0-15 hold K{0..7,16..23}, lanes 16-31 K{8..15,24..31} (n = lane%16)
        fb.lo = *reinterpret_cast<const uint4*>(&wqT[lm][K0 + half * 8]);
        fb.hi = *reinterpret_cast<const uint4*>(&wqT[lm][K0 + 16 + half * 8]);
        v16bf a = __builtin_bit_cast(v16bf, fa);
        v16bf bm = __builtin_bit_cast(v16bf, fb);
        c = __builtin_amdgcn_wmma_f32_16x16x32_bf16(false, a, false, bm,
                                                    (short)0, c, false, false);
      }
      // C layout: VGPR r -> row r (lanes 0-15) / r+8 (lanes 16-31), col = lane%16
#pragma unroll
      for (int r = 0; r < 8; ++r)
        scL[r0 + r + 8 * half][lm] = c[r] * SCALE_ + sqS[lm];
    }
    __syncthreads();

    // ---- online-softmax stats per head ----
    if (t < H_) {
      float mOld = mS[t], mc = -__builtin_inff();
      for (int r = 0; r < ROWS; ++r) mc = fmaxf(mc, scL[r][t]);
      float mNew = fmaxf(mOld, mc);
      float f = __expf(mOld - mNew);      // 0 on first chunk (mOld = -inf)
      float sum = 0.f;
      for (int r = 0; r < ROWS; ++r) {
        float w = __expf(scL[r][t] - mNew);
        awL[r][t] = w;
        sum += w;
      }
      sS[t] = sS[t] * f + sum;
      mS[t] = mNew;
      fS[t] = f;
    }
    __syncthreads();

    // ---- xbar accumulation: thread t owns cols {t, t+256, t+512} ----
#pragma unroll
    for (int h = 0; h < H_; ++h) {
      float f = fS[h];
#pragma unroll
      for (int j = 0; j < 3; ++j) acc[h][j] *= f;
    }
    for (int r = 0; r < ROWS; ++r) {
      float x0 = bf16_to_f32(xt[r][t]);
      float x1 = bf16_to_f32(xt[r][t + 256]);
      float x2 = bf16_to_f32(xt[r][t + 512]);
#pragma unroll
      for (int h = 0; h < H_; ++h) {
        float w = awL[r][h];               // LDS broadcast across lanes
        acc[h][0] = fmaf(w, x0, acc[h][0]);
        acc[h][1] = fmaf(w, x1, acc[h][1]);
        acc[h][2] = fmaf(w, x2, acc[h][2]);
      }
    }
  }

  // ---- emit split-partials: m, s, unnormalized xbar (weights exp(s - m_blk)) ----
#pragma unroll
  for (int h = 0; h < H_; ++h)
#pragma unroll
    for (int j = 0; j < 3; ++j)
      wsXb[((size_t)blk * H_ + h) * D_ + t + 256 * j] = acc[h][j];
  if (t < H_) {
    wsM[blk * H_ + t] = mS[t];
    wsS[blk * H_ + t] = sS[t];
  }
}

// ---------------------------------------------------------------------------
// Kernel 3: merge SPLITS partials per batch -> normalized xbar[b][h][d]
// ---------------------------------------------------------------------------
__global__ __launch_bounds__(256) void k3_reduce(
    const float* __restrict__ wsM, const float* __restrict__ wsS,
    const float* __restrict__ wsXb, float* __restrict__ xbarF)
{
  __shared__ float fac[SPLITS][H_];
  __shared__ float Tt[H_];
  const int b = blockIdx.x, t = threadIdx.x;
  if (t < H_) {
    float m = -__builtin_inff();
    for (int s = 0; s < SPLITS; ++s)
      m = fmaxf(m, wsM[(b * SPLITS + s) * H_ + t]);
    float T = 0.f;
    for (int s = 0; s < SPLITS; ++s) {
      float f = __expf(wsM[(b * SPLITS + s) * H_ + t] - m);
      fac[s][t] = f;
      T = fmaf(f, wsS[(b * SPLITS + s) * H_ + t], T);
    }
    Tt[t] = T;
  }
  __syncthreads();
  for (int i = t; i < H_ * D_; i += 256) {
    int h = i / D_, c = i % D_;
    float a = 0.f;
    for (int s = 0; s < SPLITS; ++s)
      a = fmaf(fac[s][h], wsXb[((size_t)(b * SPLITS + s) * H_ + h) * D_ + c], a);
    xbarF[((size_t)b * H_ + h) * D_ + c] = a / Tt[h];
  }
}

// ---------------------------------------------------------------------------
// Kernel 4 (tiny epilogue): pooled = xbar @ Wv + bv ; out = pooled @ Wproj + bproj
// ---------------------------------------------------------------------------
__global__ __launch_bounds__(256) void k4_proj(
    const float* __restrict__ xbarF, const float* __restrict__ Wkv,
    const float* __restrict__ bkv, const float* __restrict__ Wproj,
    const float* __restrict__ bproj, float* __restrict__ out)
{
  __shared__ float pooled[D_];
  const int b = blockIdx.x, t = threadIdx.x;
  for (int i = t; i < D_; i += 256) {
    int h = i / HD_;
    const float* xb = xbarF + ((size_t)b * H_ + h) * D_;
    const float* wc = Wkv + D_ + i;            // V columns of Wkv (row stride 2D)
    float s = bkv[D_ + i];
    for (int d = 0; d < D_; ++d) s = fmaf(xb[d], wc[(size_t)d * (2 * D_)], s);
    pooled[i] = s;
  }
  __syncthreads();
  for (int i = t; i < D_; i += 256) {
    float s = bproj[i];
    for (int d = 0; d < D_; ++d) s = fmaf(pooled[d], Wproj[(size_t)d * D_ + i], s);
    out[(size_t)b * D_ + i] = s;
  }
}

// ---------------------------------------------------------------------------
extern "C" void kernel_launch(void* const* d_in, const int* in_sizes, int n_in,
                              void* d_out, int out_size, void* d_ws, size_t ws_size,
                              hipStream_t stream) {
  const float* x      = (const float*)d_in[0];
  const float* latent = (const float*)d_in[1];
  const float* Wq     = (const float*)d_in[2];
  const float* bq     = (const float*)d_in[3];
  const float* Wkv    = (const float*)d_in[4];
  const float* bkv    = (const float*)d_in[5];
  const float* Wproj  = (const float*)d_in[6];
  const float* bproj  = (const float*)d_in[7];
  float* out = (float*)d_out;

  char* ws = (char*)d_ws;
  unsigned short* wsWq = (unsigned short*)ws;            // 16*768*2 = 24576 B
  float* wsSq  = (float*)(ws + 24576);                   // 64 B
  float* wsM   = (float*)(ws + 32768);                   // 256*12*4 = 12288 B
  float* wsS   = (float*)(ws + 49152);                   // 12288 B
  float* wsXb  = (float*)(ws + 65536);                   // 256*12*768*4 = 9437184 B
  float* xbarF = (float*)(ws + 65536 + 9437184);         // 32*12*768*4 = 1179648 B

  k1_prepare<<<1, 256, 0, stream>>>(latent, Wq, bq, Wkv, bkv, wsWq, wsSq);
  k2_scores_pool<<<B_ * SPLITS, 256, 0, stream>>>(x, wsWq, wsSq, wsM, wsS, wsXb);
  k3_reduce<<<B_, 256, 0, stream>>>(wsM, wsS, wsXb, xbarF);
  k4_proj<<<B_, 256, 0, stream>>>(xbarF, Wkv, bkv, Wproj, bproj, out);
}
